// GMNLayer_AT_28432683499988
// MI455X (gfx1250) — matrix operand order repaired and checked by reading, not compiled, not run on hardware
//
#include <hip/hip_runtime.h>
#include <hip/hip_bf16.h>
#include <cstdint>

#define DEVINL __device__ __forceinline__

typedef __attribute__((ext_vector_type(16))) __bf16 v16bf;
typedef __attribute__((ext_vector_type(8)))  float  v8f;

static constexpr int Bc = 2, Nc = 20000, Ec = 400000, Hc = 128;
static constexpr int HOUT = Hc + 3;           // 131
static constexpr int WAVES = 8;               // 8 waves/WG -> 2 waves per SIMD32
static constexpr int TPB = WAVES * 32;

// ---------------- edge-kernel LDS layout (bytes) ----------------
static constexpr int E_OFF_W1T  = 0;                         // u16[128][256]
static constexpr int E_OFF_W2T  = E_OFF_W1T + 128*256*2;     // u16[128][128]
static constexpr int E_OFF_CW1T = E_OFF_W2T + 128*128*2;     // u16[128][128]
static constexpr int E_OFF_W1R  = E_OFF_CW1T + 128*128*2;    // f32[128]
static constexpr int E_OFF_B1   = E_OFF_W1R + 512;
static constexpr int E_OFF_B2   = E_OFF_B1 + 512;
static constexpr int E_OFF_CB1  = E_OFF_B2 + 512;
static constexpr int E_OFF_CW2  = E_OFF_CB1 + 512;
static constexpr int E_OFF_WAVE = E_OFF_CW2 + 512;
static constexpr int E_W_HSRC = 0;        // u16[16][128]  (reused as e-buffer)
static constexpr int E_W_HTGT = 4096;     // u16[16][128]  (reused as edge_feat)
static constexpr int E_W_RAD  = 8192;     // f32[16]
static constexpr int E_W_CD   = 8256;     // f32[16][3]
static constexpr int E_W_GATE = 8448;     // f32[16]
static constexpr int E_W_ROW  = 8512;     // i32[16]
static constexpr int E_W_COL  = 8576;     // i32[16]
static constexpr int E_WAVE_BYTES = 8704;
static constexpr int EDGE_SMEM = E_OFF_WAVE + WAVES * E_WAVE_BYTES;   // ~203 KB

// ---------------- node-kernel LDS layout ----------------
static constexpr int N_OFF_W1T  = 0;                          // u16[128][384]
static constexpr int N_OFF_W2T  = N_OFF_W1T + 128*384*2;      // u16[128][128]
static constexpr int N_OFF_B1   = N_OFF_W2T + 128*128*2;
static constexpr int N_OFF_B2   = N_OFF_B1 + 512;
static constexpr int N_OFF_WAVE = N_OFF_B2 + 512;
static constexpr int N_WAVE_BYTES = 16*384*2;                 // abuf, reused for e1
static constexpr int NODE_SMEM = N_OFF_WAVE + WAVES * N_WAVE_BYTES;   // ~230 KB

// ---------------- helpers ----------------
DEVINL unsigned short f2bf(float f) {
  union { float f; unsigned u; } x; x.f = f;
  return (unsigned short)((x.u + 0x7fffu + ((x.u >> 16) & 1u)) >> 16);  // RNE
}
#if __has_builtin(__builtin_amdgcn_cvt_pk_bf16_f32)
typedef __bf16 v2bf __attribute__((ext_vector_type(2)));
DEVINL unsigned pack2(float lo, float hi) {
  v2bf r = __builtin_amdgcn_cvt_pk_bf16_f32(lo, hi);
  return __builtin_bit_cast(unsigned, r);
}
#else
DEVINL unsigned pack2(float lo, float hi) {
  return (unsigned)f2bf(lo) | ((unsigned)f2bf(hi) << 16);
}
#endif

union Frag { v16bf v; uint4 q[2]; };

// wave-local LDS sync (lockstep wave32; just order the ds traffic)
DEVINL void wsync() {
  __builtin_amdgcn_wave_barrier();
  asm volatile("s_wait_dscnt 0" ::: "memory");
  __builtin_amdgcn_wave_barrier();
}

// A-fragment: 16x32 bf16, lane = M (L&15); V0-7 hold K=kb..kb+15
// (kb = kc*32 + (lane>=16 ? 16 : 0)), rows contiguous-K in LDS.
DEVINL Frag loadA(const unsigned short* rowM, int kb) {
  Frag f;
  const uint4* p = (const uint4*)(rowM + kb);
  f.q[0] = p[0];
  f.q[1] = p[1];
  return f;
}
// B-fragment: 32x16 bf16, lane = N; lanes 0-15 hold K=kOff..+15, lanes 16-31 K=+16..+31,
// K packed 2/VGPR. Source is the [N][K]-transposed LDS weight copy (16 contiguous K).
DEVINL Frag loadB(const unsigned short* p) {
  Frag f;
  const uint4* q = (const uint4*)p;
  f.q[0] = q[0];
  f.q[1] = q[1];
  return f;
}
DEVINL v8f wmma_bf16(Frag a, Frag b, v8f c) {
  return __builtin_amdgcn_wmma_f32_16x16x32_bf16(false, a.v, false, b.v, (short)0, c,
                                                 false, false);
}
DEVINL void atomAddG(float* p, float v) {
  __hip_atomic_fetch_add(p, v, __ATOMIC_RELAXED, __HIP_MEMORY_SCOPE_AGENT);
}
DEVINL void atomAddL(float* p, float v) {
  __hip_atomic_fetch_add(p, v, __ATOMIC_RELAXED, __HIP_MEMORY_SCOPE_WORKGROUP);
}

// ---- Tensor Data Mover: load one contiguous row of 128 f32 into LDS ----
// D# per CDNA5 ISA: group0 {count=1, lds_addr, global_addr, type=2},
// group1 {data_size=4B, tensor_dim0=128, tensor_dim1=1, tile_dim0=128, tile_dim1=1,
//         tensor_dim0_stride=128}. Groups 2/3 zero (<=2D tensor).
#if __has_builtin(__builtin_amdgcn_tensor_load_to_lds)
#define HAVE_TDM 1
typedef unsigned int tdm_v4u __attribute__((ext_vector_type(4)));
typedef int tdm_v8i __attribute__((ext_vector_type(8)));
typedef int tdm_v4i __attribute__((ext_vector_type(4)));
DEVINL void tdm_load_row128(unsigned ldsByteOff, const float* gptr) {
  unsigned long long ga = (unsigned long long)(uintptr_t)gptr;
  tdm_v4u g0;
  g0[0] = 1u;                                               // count=1 (user mode)
  g0[1] = ldsByteOff;                                       // lds_addr (bytes)
  g0[2] = (unsigned)(ga & 0xffffffffu);                     // global_addr[31:0]
  g0[3] = (unsigned)((ga >> 32) & 0x01ffffffu) | (2u << 30); // addr[56:32] | type=2
  tdm_v8i g1;
  g1[0] = 0x20000;                 // workgroup_mask=0, data_size=2 (4 bytes)
  g1[1] = (int)(128u << 16);       // tensor_dim0 = 128 (bits 79:48, low16 here)
  g1[2] = (int)(1u << 16);         // tensor_dim1 = 1   (bits 111:80, low16 here)
  g1[3] = (int)(128u << 16);       // tile_dim0 = 128   (bits 127:112)
  g1[4] = 1;                       // tile_dim1 = 1
  g1[5] = 128;                     // tensor_dim0_stride = 128
  g1[6] = 0;
  g1[7] = 0;
  tdm_v4i z = {0, 0, 0, 0};
#if defined(__clang_major__) && __clang_major__ >= 23
  tdm_v8i z8 = {0, 0, 0, 0, 0, 0, 0, 0};
  __builtin_amdgcn_tensor_load_to_lds(g0, g1, z, z, z8, 0);
#else
  __builtin_amdgcn_tensor_load_to_lds(g0, g1, z, z, 0);
#endif
}
#else
#define HAVE_TDM 0
#endif

// ================= edge kernel =================
__global__ __launch_bounds__(TPB)
void egnn_edge_kernel(const float* __restrict__ h, const float* __restrict__ coord,
                      const int* __restrict__ erow, const int* __restrict__ ecol,
                      const float* __restrict__ ew1, const float* __restrict__ eb1,
                      const float* __restrict__ ew2, const float* __restrict__ eb2,
                      const float* __restrict__ cw1, const float* __restrict__ cb1,
                      const float* __restrict__ cw2,
                      float* __restrict__ agg, float* __restrict__ aggT) {
  extern __shared__ __align__(16) unsigned char smem[];
  unsigned short* w1T  = (unsigned short*)(smem + E_OFF_W1T);
  unsigned short* w2T  = (unsigned short*)(smem + E_OFF_W2T);
  unsigned short* cw1T = (unsigned short*)(smem + E_OFF_CW1T);
  float* w1r  = (float*)(smem + E_OFF_W1R);
  float* b1s  = (float*)(smem + E_OFF_B1);
  float* b2s  = (float*)(smem + E_OFF_B2);
  float* cb1s = (float*)(smem + E_OFF_CB1);
  float* cw2s = (float*)(smem + E_OFF_CW2);

  const int tid = threadIdx.x;
#if HAVE_TDM
  // wave 0 DMAs the f32 vectors (biases, W1 radial row, cw2) straight into LDS
  if (tid < 32) {
    tdm_load_row128(E_OFF_W1R, ew1 + 256 * Hc);
    tdm_load_row128(E_OFF_B1, eb1);
    tdm_load_row128(E_OFF_B2, eb2);
    tdm_load_row128(E_OFF_CB1, cb1);
    tdm_load_row128(E_OFF_CW2, cw2);
    __builtin_amdgcn_s_wait_tensorcnt(0);
  }
#else
  for (int i = tid; i < 128; i += TPB) {
    w1r[i]  = ew1[256 * Hc + i];
    b1s[i]  = eb1[i];
    b2s[i]  = eb2[i];
    cb1s[i] = cb1[i];
    cw2s[i] = cw2[i];
  }
#endif
  // stage transposed bf16 weights once per WG
  for (int idx = tid; idx < 128 * 256; idx += TPB) {
    int n = idx >> 8, k = idx & 255;
    w1T[n * 256 + k] = f2bf(ew1[k * Hc + n]);
  }
  for (int idx = tid; idx < 128 * 128; idx += TPB) {
    int n = idx >> 7, k = idx & 127;
    w2T[n * 128 + k]  = f2bf(ew2[k * Hc + n]);
    cw1T[n * 128 + k] = f2bf(cw1[k * Hc + n]);
  }
  __syncthreads();

  const int wave = tid >> 5;
  const int L = tid & 31;
  unsigned char* ws = smem + E_OFF_WAVE + wave * E_WAVE_BYTES;
  unsigned short* hsrc = (unsigned short*)(ws + E_W_HSRC);
  unsigned short* htgt = (unsigned short*)(ws + E_W_HTGT);
  float* rad  = (float*)(ws + E_W_RAD);
  float* cd   = (float*)(ws + E_W_CD);
  float* gate = (float*)(ws + E_W_GATE);
  int* rowi = (int*)(ws + E_W_ROW);
  int* coli = (int*)(ws + E_W_COL);

  const int n_ = L & 15;
  const int base = (L < 16) ? 0 : 8;
  const int khalf = (L >> 4) << 4;             // 0 or 16 for B frags
  const int aoff  = (L >= 16) ? 16 : 0;        // A frag K sub-offset
  const v8f vzero = {0, 0, 0, 0, 0, 0, 0, 0};

  const int tilesPerBatch = Ec / 16;
  const int nTiles = Bc * tilesPerBatch;
  const int stride = gridDim.x * WAVES;
  for (int tile = blockIdx.x * WAVES + wave; tile < nTiles; tile += stride) {
    const int b = tile / tilesPerBatch;
    const int e0 = (tile - b * tilesPerBatch) * 16;
    const float* hb = h + (size_t)b * Nc * Hc;
    const float* cb = coord + (size_t)b * Nc * 3;

    // prefetch next tile's edge indices into cache
    {
      int nxt = tile + stride;
      if (L == 0 && nxt < nTiles) {
        int nb = nxt / tilesPerBatch;
        int ne0 = (nxt - nb * tilesPerBatch) * 16;
        __builtin_prefetch(&erow[ne0], 0, 0);
        __builtin_prefetch(&ecol[ne0], 0, 0);
      }
    }

    if (L < 16) {
      int r = erow[e0 + L], c = ecol[e0 + L];
      rowi[L] = r; coli[L] = c;
      float dx = cb[r * 3 + 0] - cb[c * 3 + 0];
      float dy = cb[r * 3 + 1] - cb[c * 3 + 1];
      float dz = cb[r * 3 + 2] - cb[c * 3 + 2];
      cd[L * 3 + 0] = dx; cd[L * 3 + 1] = dy; cd[L * 3 + 2] = dz;
      rad[L] = dx * dx + dy * dy + dz * dz;
      gate[L] = 0.0f;
    }
    wsync();

    // gather h[src]/h[tgt] rows -> bf16 LDS (2 lanes per row)
    {
      int j = L >> 1, half = L & 1, f0 = half * 64;
      const float4* hs4 = (const float4*)(hb + (size_t)rowi[j] * Hc + f0);
      const float4* ht4 = (const float4*)(hb + (size_t)coli[j] * Hc + f0);
      uint4* ds = (uint4*)(hsrc + j * 128 + f0);
      uint4* dt = (uint4*)(htgt + j * 128 + f0);
#pragma unroll
      for (int i = 0; i < 8; i++) {
        float4 a = hs4[2 * i], bq = hs4[2 * i + 1];
        ds[i] = make_uint4(pack2(a.x, a.y), pack2(a.z, a.w),
                           pack2(bq.x, bq.y), pack2(bq.z, bq.w));
        float4 c0 = ht4[2 * i], c1 = ht4[2 * i + 1];
        dt[i] = make_uint4(pack2(c0.x, c0.y), pack2(c0.z, c0.w),
                           pack2(c1.x, c1.y), pack2(c1.z, c1.w));
      }
    }
    wsync();

    const unsigned short* aRowS = hsrc + n_ * 128;
    const unsigned short* aRowT = htgt + n_ * 128;

    // ---- GEMM1: e = relu([src|tgt] @ W1[0:256] + radial*W1[256] + b1) ----
    v8f acc[8];
#pragma unroll
    for (int t = 0; t < 8; t++) acc[t] = vzero;
#pragma unroll
    for (int kc = 0; kc < 4; kc++) {
      Frag fa = loadA(aRowS, kc * 32 + aoff);
#pragma unroll
      for (int nt = 0; nt < 8; nt++) {
        Frag fb = loadB(w1T + (size_t)(nt * 16 + n_) * 256 + kc * 32 + khalf);
        acc[nt] = wmma_bf16(fa, fb, acc[nt]);
      }
    }
#pragma unroll
    for (int kc = 0; kc < 4; kc++) {
      Frag fa = loadA(aRowT, kc * 32 + aoff);
#pragma unroll
      for (int nt = 0; nt < 8; nt++) {
        Frag fb = loadB(w1T + (size_t)(nt * 16 + n_) * 256 + 128 + kc * 32 + khalf);
        acc[nt] = wmma_bf16(fa, fb, acc[nt]);
      }
    }
    float rv8[8];
#pragma unroll
    for (int v = 0; v < 8; v++) rv8[v] = rad[base + v];
#pragma unroll
    for (int nt = 0; nt < 8; nt++) {
      float wrn = w1r[nt * 16 + n_];
      float bn = b1s[nt * 16 + n_];
#pragma unroll
      for (int v = 0; v < 8; v++) {
        float x = acc[nt][v] + rv8[v] * wrn + bn;
        acc[nt][v] = x > 0.f ? x : 0.f;
      }
    }
    wsync();
    // store e (bf16) over hsrc for GEMM2 A frags
#pragma unroll
    for (int nt = 0; nt < 8; nt++)
#pragma unroll
      for (int v = 0; v < 8; v++)
        hsrc[(v + base) * 128 + nt * 16 + n_] = f2bf(acc[nt][v]);
    wsync();

    // ---- GEMM2: edge_feat = relu(e @ W2 + b2) ----
#pragma unroll
    for (int t = 0; t < 8; t++) acc[t] = vzero;
#pragma unroll
    for (int kc = 0; kc < 4; kc++) {
      Frag fa = loadA(aRowS, kc * 32 + aoff);
#pragma unroll
      for (int nt = 0; nt < 8; nt++) {
        Frag fb = loadB(w2T + (size_t)(nt * 16 + n_) * 128 + kc * 32 + khalf);
        acc[nt] = wmma_bf16(fa, fb, acc[nt]);
      }
    }
    int rix[8];
#pragma unroll
    for (int v = 0; v < 8; v++) rix[v] = rowi[base + v];
    wsync();
#pragma unroll
    for (int nt = 0; nt < 8; nt++) {
      float bn = b2s[nt * 16 + n_];
#pragma unroll
      for (int v = 0; v < 8; v++) {
        float x = acc[nt][v] + bn;
        x = x > 0.f ? x : 0.f;
        htgt[(v + base) * 128 + nt * 16 + n_] = f2bf(x);  // keep for GEMM3
        atomAddG(&agg[((size_t)b * Nc + rix[v]) * Hc + nt * 16 + n_], x);
      }
    }
    wsync();

    // ---- GEMM3: gate = (relu(edge_feat @ CW1 + cb1)) @ cw2 ----
#pragma unroll
    for (int t = 0; t < 8; t++) acc[t] = vzero;
#pragma unroll
    for (int kc = 0; kc < 4; kc++) {
      Frag fa = loadA(aRowT, kc * 32 + aoff);
#pragma unroll
      for (int nt = 0; nt < 8; nt++) {
        Frag fb = loadB(cw1T + (size_t)(nt * 16 + n_) * 128 + kc * 32 + khalf);
        acc[nt] = wmma_bf16(fa, fb, acc[nt]);
      }
    }
    float p[8] = {0, 0, 0, 0, 0, 0, 0, 0};
#pragma unroll
    for (int nt = 0; nt < 8; nt++) {
      float bn = cb1s[nt * 16 + n_];
      float c2 = cw2s[nt * 16 + n_];
#pragma unroll
      for (int v = 0; v < 8; v++) {
        float x = acc[nt][v] + bn;
        x = x > 0.f ? x : 0.f;
        p[v] += x * c2;
      }
    }
#pragma unroll
    for (int v = 0; v < 8; v++) atomAddL(&gate[base + v], p[v]);
    wsync();
    if (L < 16) {
      float g = gate[L];
      int r = rowi[L];
      float* at = aggT + ((size_t)b * Nc + r) * 3;
      atomAddG(&at[0], cd[L * 3 + 0] * g);
      atomAddG(&at[1], cd[L * 3 + 1] * g);
      atomAddG(&at[2], cd[L * 3 + 2] * g);
    }
    wsync();
  }
}

// ================= node kernel =================
__global__ __launch_bounds__(TPB)
void egnn_node_kernel(const float* __restrict__ h, const float* __restrict__ others,
                      const float* __restrict__ agg,
                      const float* __restrict__ nw1, const float* __restrict__ nb1,
                      const float* __restrict__ nw2, const float* __restrict__ nb2,
                      float* __restrict__ out) {
  extern __shared__ __align__(16) unsigned char smem[];
  unsigned short* w1T = (unsigned short*)(smem + N_OFF_W1T);
  unsigned short* w2T = (unsigned short*)(smem + N_OFF_W2T);
  float* b1s = (float*)(smem + N_OFF_B1);
  float* b2s = (float*)(smem + N_OFF_B2);

  const int tid = threadIdx.x;
#if HAVE_TDM
  if (tid < 32) {
    tdm_load_row128(N_OFF_B1, nb1);
    tdm_load_row128(N_OFF_B2, nb2);
    __builtin_amdgcn_s_wait_tensorcnt(0);
  }
#else
  for (int i = tid; i < 128; i += TPB) { b1s[i] = nb1[i]; b2s[i] = nb2[i]; }
#endif
  for (int idx = tid; idx < 128 * 384; idx += TPB) {
    int n = idx / 384, k = idx - n * 384;
    w1T[idx] = f2bf(nw1[k * Hc + n]);       // [N][K] transposed
  }
  for (int idx = tid; idx < 128 * 128; idx += TPB) {
    int n = idx >> 7, k = idx & 127;
    w2T[idx] = f2bf(nw2[k * Hc + n]);       // idx == n*128+k
  }
  __syncthreads();

  const int wave = tid >> 5;
  const int L = tid & 31;
  unsigned short* abuf = (unsigned short*)(smem + N_OFF_WAVE + wave * N_WAVE_BYTES);

  const int n_ = L & 15;
  const int base = (L < 16) ? 0 : 8;
  const int khalf = (L >> 4) << 4;
  const int aoff  = (L >= 16) ? 16 : 0;
  const v8f vzero = {0, 0, 0, 0, 0, 0, 0, 0};

  const int tilesPerBatch = Nc / 16;
  const int nTiles = Bc * tilesPerBatch;
  for (int tile = blockIdx.x * WAVES + wave; tile < nTiles; tile += gridDim.x * WAVES) {
    const int b = tile / tilesPerBatch;
    const int n0 = (tile - b * tilesPerBatch) * 16;

    // stage a = [others | h | agg] rows as bf16 (2 lanes per row)
    {
      int j = L >> 1, half = L & 1, f0 = half * 192;
      size_t roff = (size_t)b * Nc + n0 + j;
      const float* orow = others + roff * Hc;
      const float* hrow = h + roff * Hc;
      const float* grow = agg + roff * Hc;
#pragma unroll
      for (int g = 0; g < 24; g++) {
        int f = f0 + g * 8;
        const float* src = (f < 128) ? (orow + f)
                         : (f < 256) ? (hrow + (f - 128))
                                     : (grow + (f - 256));
        float4 a = ((const float4*)src)[0], bq = ((const float4*)src)[1];
        *(uint4*)(abuf + j * 384 + f) = make_uint4(pack2(a.x, a.y), pack2(a.z, a.w),
                                                   pack2(bq.x, bq.y), pack2(bq.z, bq.w));
      }
    }
    wsync();

    // ---- GEMM1: e1 = relu(a @ NW1 + nb1), K = 384 ----
    v8f acc[8];
#pragma unroll
    for (int t = 0; t < 8; t++) acc[t] = vzero;
    const unsigned short* aRow = abuf + n_ * 384;
#pragma unroll
    for (int kc = 0; kc < 12; kc++) {
      Frag fa = loadA(aRow, kc * 32 + aoff);
#pragma unroll
      for (int nt = 0; nt < 8; nt++) {
        Frag fb = loadB(w1T + (size_t)(nt * 16 + n_) * 384 + kc * 32 + khalf);
        acc[nt] = wmma_bf16(fa, fb, acc[nt]);
      }
    }
#pragma unroll
    for (int nt = 0; nt < 8; nt++) {
      float bn = b1s[nt * 16 + n_];
#pragma unroll
      for (int v = 0; v < 8; v++) {
        float x = acc[nt][v] + bn;
        acc[nt][v] = x > 0.f ? x : 0.f;
      }
    }
    wsync();
    // store e1 bf16 compact (stride 128) over abuf
#pragma unroll
    for (int nt = 0; nt < 8; nt++)
#pragma unroll
      for (int v = 0; v < 8; v++)
        abuf[(v + base) * 128 + nt * 16 + n_] = f2bf(acc[nt][v]);
    wsync();

    // ---- GEMM2: o = e1 @ NW2 + nb2;  out = h + o ----
#pragma unroll
    for (int t = 0; t < 8; t++) acc[t] = vzero;
    const unsigned short* eRow = abuf + n_ * 128;
#pragma unroll
    for (int kc = 0; kc < 4; kc++) {
      Frag fa = loadA(eRow, kc * 32 + aoff);
#pragma unroll
      for (int nt = 0; nt < 8; nt++) {
        Frag fb = loadB(w2T + (size_t)(nt * 16 + n_) * 128 + kc * 32 + khalf);
        acc[nt] = wmma_bf16(fa, fb, acc[nt]);
      }
    }
#pragma unroll
    for (int nt = 0; nt < 8; nt++) {
      int col = nt * 16 + n_;
      float bn = b2s[col];
#pragma unroll
      for (int v = 0; v < 8; v++) {
        size_t node = (size_t)b * Nc + n0 + v + base;
        float hv = h[node * Hc + col];
        out[node * HOUT + col] = hv + acc[nt][v] + bn;
      }
    }
    wsync();
  }
}

// ================= small kernels =================
__global__ void egnn_cnt_kernel(const int* __restrict__ erow, float* __restrict__ cnt) {
  for (int i = blockIdx.x * blockDim.x + threadIdx.x; i < Ec; i += gridDim.x * blockDim.x)
    __hip_atomic_fetch_add(&cnt[erow[i]], 1.0f, __ATOMIC_RELAXED, __HIP_MEMORY_SCOPE_AGENT);
}

__global__ void egnn_coord_kernel(const float* __restrict__ coord,
                                  const float* __restrict__ aggT,
                                  const float* __restrict__ cnt,
                                  float* __restrict__ out) {
  int t = blockIdx.x * blockDim.x + threadIdx.x;
  if (t >= Bc * Nc) return;
  int nidx = t % Nc;
  float c = cnt[nidx];
  c = c < 1.f ? 1.f : c;
  float inv = 1.f / c;
#pragma unroll
  for (int x = 0; x < 3; x++)
    out[(size_t)t * HOUT + Hc + x] = coord[(size_t)t * 3 + x] + aggT[(size_t)t * 3 + x] * inv;
}

// ================= launch =================
extern "C" void kernel_launch(void* const* d_in, const int* in_sizes, int n_in,
                              void* d_out, int out_size, void* d_ws, size_t ws_size,
                              hipStream_t stream) {
  (void)in_sizes; (void)n_in; (void)out_size; (void)ws_size;
  const float* h      = (const float*)d_in[0];
  const float* coord  = (const float*)d_in[1];
  const float* others = (const float*)d_in[2];
  const int*   eidx   = (const int*)d_in[3];
  const float* ew1 = (const float*)d_in[4];
  const float* eb1 = (const float*)d_in[5];
  const float* ew2 = (const float*)d_in[6];
  const float* eb2 = (const float*)d_in[7];
  const float* nw1 = (const float*)d_in[8];
  const float* nb1 = (const float*)d_in[9];
  const float* nw2 = (const float*)d_in[10];
  const float* nb2 = (const float*)d_in[11];
  const float* cw1 = (const float*)d_in[12];
  const float* cb1 = (const float*)d_in[13];
  const float* cw2 = (const float*)d_in[14];
  float* out = (float*)d_out;

  float* agg  = (float*)d_ws;                    // B*N*H
  float* aggT = agg + (size_t)Bc * Nc * Hc;      // B*N*3
  float* cnt  = aggT + (size_t)Bc * Nc * 3;      // N
  size_t zbytes = ((size_t)Bc * Nc * Hc + (size_t)Bc * Nc * 3 + Nc) * sizeof(float);
  hipMemsetAsync(d_ws, 0, zbytes, stream);

  const int* erow = eidx;
  const int* ecol = eidx + Ec;

  hipLaunchKernelGGL(egnn_cnt_kernel, dim3(512), dim3(256), 0, stream, erow, cnt);
  hipLaunchKernelGGL(egnn_edge_kernel, dim3(512), dim3(TPB), EDGE_SMEM, stream,
                     h, coord, erow, ecol, ew1, eb1, ew2, eb2, cw1, cb1, cw2, agg, aggT);
  hipLaunchKernelGGL(egnn_node_kernel, dim3(160), dim3(TPB), NODE_SMEM, stream,
                     h, others, agg, nw1, nb1, nw2, nb2, out);
  hipLaunchKernelGGL(egnn_coord_kernel, dim3((Bc * Nc + 255) / 256), dim3(256), 0, stream,
                     coord, aggT, cnt, out);
}
